// Network_33672543601392
// MI455X (gfx1250) — matrix operand order repaired and checked by reading, not compiled, not run on hardware
//
#include <hip/hip_runtime.h>
#include <hip/hip_bf16.h>
#include <math.h>

// MemN2N + LSTM encoder for MI455X (gfx1250, wave32, WMMA).
// All GEMMs use v_wmma_f32_16x16x32_f16 (f16 in, f32 accumulate).

typedef __attribute__((ext_vector_type(16))) _Float16 v16h;
typedef __attribute__((ext_vector_type(8)))  float    v8f;

#define WPB 8  // waves per block (256 threads)

// ---------------------------------------------------------------------------
// WMMA K-loop helper: one 16x16 f32 tile of A(row-major, K cols) x Bmat^T,
// where logical B[k][n] = Bmat[n][k] (Bmat row-major, K cols).
// Arow / Brow are this lane's row base pointers (A + aRow*K, Bmat + bRow*K).
// A and B share contiguous k order: lane half h holds k = 16h..16h+15 of each 32-k block.
// ---------------------------------------------------------------------------
__device__ __forceinline__ v8f wmma_k_loop(const _Float16* Arow,
                                           const _Float16* Brow, int K)
{
    int lane = threadIdx.x & 31;
    int aOff = (lane & 16) ? 16 : 0;   // A: K 0-15 vs 16-31
    int bOff = (lane & 16) ? 16 : 0;   // B: K 0-15 vs 16-31
    v8f acc = {};
    for (int kb = 0; kb < K; kb += 32) {
        v16h af, bf;
        const _Float16* pa = Arow + kb + aOff;
#pragma unroll
        for (int i = 0; i < 16; ++i) af[i] = pa[i];
        const _Float16* pb = Brow + kb + bOff;
#pragma unroll
        for (int i = 0; i < 16; ++i) bf[i] = pb[i];
        acc = __builtin_amdgcn_wmma_f32_16x16x32_f16(false, af, false, bf,
                                                     (short)0, acc, false, false);
    }
    return acc;
}

// D[M][N] (f32, optional f16 mirror) = A[M][K] * Bmat[N][K]^T. One tile/wave.
__global__ void gemm_abt(const _Float16* __restrict__ A,
                         const _Float16* __restrict__ Bm,
                         float* __restrict__ Dout, _Float16* __restrict__ Dh,
                         int M, int N, int K)
{
    int wave   = threadIdx.x >> 5;
    int tile   = blockIdx.x * WPB + wave;
    int tilesN = N >> 4;
    if (tile >= (M >> 4) * tilesN) return;      // wave-uniform: EXEC stays full
    int tm = tile / tilesN, tn = tile - tm * tilesN;
    int lane = threadIdx.x & 31;
    int aRow = tm * 16 + (lane & 15);
    int bRow = tn * 16 + (lane & 15);
    v8f acc = wmma_k_loop(A + (size_t)aRow * K, Bm + (size_t)bRow * K, K);
    int m0 = tm * 16 + ((lane & 16) ? 8 : 0);
    int n  = tn * 16 + (lane & 15);
#pragma unroll
    for (int r = 0; r < 8; ++r) {
        float v = acc[r];
        Dout[(size_t)(m0 + r) * N + n] = v;
        if (Dh) Dh[(size_t)(m0 + r) * N + n] = (_Float16)v;
    }
}

// ---------------------------------------------------------------------------
// Gather+sum memory embeddings: M[h][n][d] = sum_w A_emb[h][memory[n][w]][d]
// Also writes transposed, K-padded Mt[h][d][n] (cols 2000..2015 pre-zeroed).
// ---------------------------------------------------------------------------
__global__ void gather_M(const float* __restrict__ A_emb,
                         const int* __restrict__ memory,
                         _Float16* __restrict__ M, _Float16* __restrict__ Mt)
{
    int hop = blockIdx.x / 2000;
    int n   = blockIdx.x % 2000;
    __shared__ int idx[50];
    if (threadIdx.x < 50) idx[threadIdx.x] = memory[n * 50 + threadIdx.x];
    __syncthreads();
    int d = threadIdx.x;                          // 256 threads = 256 dims
    const float* base = A_emb + (size_t)hop * 50000 * 256;
    float s = 0.f;
    for (int w = 0; w < 50; ++w) {
        if (w + 6 < 50)
            __builtin_prefetch(base + (size_t)idx[w + 6] * 256 + d, 0, 1);
        s += base[(size_t)idx[w] * 256 + d];
    }
    _Float16 hv = (_Float16)s;
    M [((size_t)hop * 2000 + n) * 256 + d]  = hv;
    Mt[((size_t)hop * 256  + d) * 2016 + n] = hv;
}

__global__ void embed_q(const float* __restrict__ B_emb,
                        const int* __restrict__ ques, _Float16* __restrict__ xq)
{
    int b = blockIdx.x / 50, t = blockIdx.x % 50;
    int tok = ques[b * 50 + t];
    xq[((size_t)b * 50 + t) * 256 + threadIdx.x] =
        (_Float16)B_emb[(size_t)tok * 256 + threadIdx.x];
}

__global__ void embed_a(const float* __restrict__ C_emb,
                        const int* __restrict__ ach,
                        float* __restrict__ a, _Float16* __restrict__ ah)
{
    int bk = blockIdx.x;                          // 64*4 rows
    float s = 0.f;
    for (int w = 0; w < 10; ++w)
        s += C_emb[(size_t)ach[bk * 10 + w] * 256 + threadIdx.x];
    a [bk * 256 + threadIdx.x] = s;
    ah[bk * 256 + threadIdx.x] = (_Float16)s;
}

// ---------------------------------------------------------------------------
// One LSTM step, fully fused. Block bi owns batches 16bi..16bi+15 and all 512
// gate outputs: g[16][512] = [x_t | h] (16x384) * WcT[512][384]^T + bias.
// ---------------------------------------------------------------------------
__global__ void lstm_step(const _Float16* __restrict__ xq,   // [64][50][256]
                          const _Float16* __restrict__ WcT,  // [512][384]
                          const float* __restrict__ bias,    // [512]
                          float* __restrict__ h, float* __restrict__ c, int t)
{
    __shared__ _Float16 Xs[16 * 384];
    __shared__ float    Gs[16 * 512];
    int b0 = blockIdx.x * 16;
    for (int i = threadIdx.x; i < 16 * 384; i += 256) {
        int row = i / 384, col = i % 384;
        Xs[i] = (col < 256)
                    ? xq[((size_t)(b0 + row) * 50 + t) * 256 + col]
                    : (_Float16)h[(b0 + row) * 128 + (col - 256)];
    }
    __syncthreads();
    int wave = threadIdx.x >> 5, lane = threadIdx.x & 31;
    int aRow = lane & 15;
    for (int nt = wave; nt < 32; nt += 8) {       // 4 N-tiles per wave
        int bRow = nt * 16 + (lane & 15);
        v8f acc = wmma_k_loop(Xs + aRow * 384, WcT + (size_t)bRow * 384, 384);
        int m0 = (lane & 16) ? 8 : 0;
        int n  = nt * 16 + (lane & 15);
#pragma unroll
        for (int r = 0; r < 8; ++r) Gs[(m0 + r) * 512 + n] = acc[r] + bias[n];
    }
    __syncthreads();
    for (int i = threadIdx.x; i < 16 * 128; i += 256) {   // gate order i,f,g,o
        int row = i / 128, hh = i % 128;
        float gi = Gs[row * 512 + hh];
        float gf = Gs[row * 512 + 128 + hh];
        float gg = Gs[row * 512 + 256 + hh];
        float go = Gs[row * 512 + 384 + hh];
        int   gx = (b0 + row) * 128 + hh;
        float si = 1.f / (1.f + __expf(-gi));
        float sf = 1.f / (1.f + __expf(-gf));
        float so = 1.f / (1.f + __expf(-go));
        float cn = sf * c[gx] + si * tanhf(gg);
        c[gx] = cn;
        h[gx] = so * tanhf(cn);
    }
}

__global__ void finalize_u(const float* __restrict__ h, const float* __restrict__ c,
                           float* __restrict__ u, _Float16* __restrict__ uh)
{
    int b = blockIdx.x, d = threadIdx.x;
    float v = (d < 128) ? h[b * 128 + d] : c[b * 128 + (d - 128)];
    u [b * 256 + d] = v;
    uh[b * 256 + d] = (_Float16)v;
}

// Row softmax over 2000 slots; writes f16 probs padded to 2016 with zeros.
__global__ void softmax_rows(const float* __restrict__ S, _Float16* __restrict__ P)
{
    int row = blockIdx.x;
    const float* s = S + (size_t)row * 2000;
    __shared__ float red[8], red2[8];
    float mx = -1e30f;
    for (int i = threadIdx.x; i < 2000; i += 256) mx = fmaxf(mx, s[i]);
    for (int o = 16; o; o >>= 1) mx = fmaxf(mx, __shfl_xor(mx, o, 32));
    if ((threadIdx.x & 31) == 0) red[threadIdx.x >> 5] = mx;
    __syncthreads();
    if (threadIdx.x == 0) {
        float m = red[0];
        for (int i = 1; i < 8; ++i) m = fmaxf(m, red[i]);
        red[0] = m;
    }
    __syncthreads();
    mx = red[0];
    float sum = 0.f;
    for (int i = threadIdx.x; i < 2000; i += 256) sum += __expf(s[i] - mx);
    for (int o = 16; o; o >>= 1) sum += __shfl_xor(sum, o, 32);
    if ((threadIdx.x & 31) == 0) red2[threadIdx.x >> 5] = sum;
    __syncthreads();
    if (threadIdx.x == 0) {
        float tt = 0.f;
        for (int i = 0; i < 8; ++i) tt += red2[i];
        red2[0] = tt;
    }
    __syncthreads();
    float inv = 1.f / red2[0];
    for (int i = threadIdx.x; i < 2016; i += 256)
        P[(size_t)row * 2016 + i] =
            (i < 2000) ? (_Float16)(__expf(s[i] - mx) * inv) : (_Float16)0.f;
}

// u += o_q (o comes in transposed [256][64]); also emit o_q as f16 row-major.
__global__ void add_u(float* __restrict__ u, _Float16* __restrict__ uh,
                      const float* __restrict__ oT, _Float16* __restrict__ oh)
{
    int b = blockIdx.x, d = threadIdx.x;
    float o  = oT[(size_t)d * 64 + b];
    float un = u[b * 256 + d] + o;
    u [b * 256 + d] = un;
    uh[b * 256 + d] = (_Float16)un;
    oh[b * 256 + d] = (_Float16)o;
}

__global__ void add_a(float* __restrict__ a, _Float16* __restrict__ ah,
                      const float* __restrict__ oT, _Float16* __restrict__ oh)
{
    int bk = blockIdx.x, d = threadIdx.x;
    float o  = oT[(size_t)d * 256 + bk];
    float an = a[bk * 256 + d] + o;
    a [bk * 256 + d] = an;
    ah[bk * 256 + d] = (_Float16)an;
    oh[bk * 256 + d] = (_Float16)o;
}

// pred[b][k] = r[b]·o_a[b][k]; log_softmax over k=4. One block per b, wave per k.
__global__ void predict(const float* __restrict__ r, const _Float16* __restrict__ oah,
                        float* __restrict__ out)
{
    int b = blockIdx.x;
    int k = threadIdx.x >> 5, lane = threadIdx.x & 31;
    float p = 0.f;
    for (int m = lane; m < 256; m += 32)
        p += r[b * 256 + m] * (float)oah[((size_t)b * 4 + k) * 256 + m];
    for (int o = 16; o; o >>= 1) p += __shfl_xor(p, o, 32);
    __shared__ float pv[4];
    if (lane == 0) pv[k] = p;
    __syncthreads();
    if (threadIdx.x < 4) {
        float mx = fmaxf(fmaxf(pv[0], pv[1]), fmaxf(pv[2], pv[3]));
        float lse = logf(__expf(pv[0] - mx) + __expf(pv[1] - mx) +
                         __expf(pv[2] - mx) + __expf(pv[3] - mx));
        out[b * 4 + threadIdx.x] = pv[threadIdx.x] - mx - lse;
    }
}

// ------------------------------ init utils --------------------------------
__global__ void zero_f32(float* p, int n)
{ int i = blockIdx.x * 256 + threadIdx.x; if (i < n) p[i] = 0.f; }

__global__ void zero_h16(_Float16* p, int n)
{ int i = blockIdx.x * 256 + threadIdx.x; if (i < n) p[i] = (_Float16)0.f; }

__global__ void cvt_h16(_Float16* d, const float* s, int n)
{ int i = blockIdx.x * 256 + threadIdx.x; if (i < n) d[i] = (_Float16)s[i]; }

__global__ void transpose_h16(_Float16* d, const float* s, int R, int C)
{
    int i = blockIdx.x * 256 + threadIdx.x;
    if (i < R * C) { int r = i / C, c = i % C; d[c * R + r] = (_Float16)s[r * C + c]; }
}

__global__ void build_WcT(_Float16* d, const float* wih, const float* whh)
{
    int i = blockIdx.x * 256 + threadIdx.x;           // 512*384
    if (i < 512 * 384) {
        int j = i / 384, k = i % 384;
        d[i] = (_Float16)((k < 256) ? wih[j * 256 + k] : whh[j * 128 + (k - 256)]);
    }
}

__global__ void bias_sum(float* b, const float* bi, const float* bh)
{ int i = blockIdx.x * 256 + threadIdx.x; if (i < 512) b[i] = bi[i] + bh[i]; }

// ---------------------------------------------------------------------------
extern "C" void kernel_launch(void* const* d_in, const int* in_sizes, int n_in,
                              void* d_out, int out_size, void* d_ws, size_t ws_size,
                              hipStream_t stream)
{
    (void)in_sizes; (void)n_in; (void)out_size; (void)ws_size;
    const int*   memory = (const int*)  d_in[0];
    const int*   ques   = (const int*)  d_in[1];
    const int*   ach    = (const int*)  d_in[2];
    const float* A_emb  = (const float*)d_in[3];
    const float* B_emb  = (const float*)d_in[4];
    const float* C_emb  = (const float*)d_in[5];
    const float* U      = (const float*)d_in[6];
    const float* V      = (const float*)d_in[7];
    const float* W      = (const float*)d_in[8];
    const float* W_ih   = (const float*)d_in[9];
    const float* W_hh   = (const float*)d_in[10];
    const float* b_ih   = (const float*)d_in[11];
    const float* b_hh   = (const float*)d_in[12];
    float* out = (float*)d_out;

    char* base = (char*)d_ws; size_t off = 0;
    auto alloc = [&](size_t bytes) -> void* {
        off = (off + 255) & ~(size_t)255;
        void* p = base + off; off += bytes; return p;
    };
    _Float16* M_h   = (_Float16*)alloc((size_t)3 * 2000 * 256 * 2);
    _Float16* Mt_h  = (_Float16*)alloc((size_t)3 * 256 * 2016 * 2);
    _Float16* xq_h  = (_Float16*)alloc((size_t)64 * 50 * 256 * 2);
    _Float16* WcT_h = (_Float16*)alloc((size_t)512 * 384 * 2);
    float*    bias  = (float*)   alloc(512 * 4);
    float*    hbuf  = (float*)   alloc((size_t)2 * 64 * 128 * 4);
    float*    cbuf  = hbuf + 64 * 128;
    _Float16* U_h   = (_Float16*)alloc(65536 * 2);
    _Float16* V_h   = (_Float16*)alloc(65536 * 2);
    _Float16* Wt_h  = (_Float16*)alloc(65536 * 2);
    float*    u_f   = (float*)   alloc(64 * 256 * 4);
    _Float16* u_h   = (_Float16*)alloc(64 * 256 * 2);
    float*    a_f   = (float*)   alloc(256 * 256 * 4);
    _Float16* a_h   = (_Float16*)alloc(256 * 256 * 2);
    float*    tq_f  = (float*)   alloc(64 * 256 * 4);
    _Float16* tq_h  = (_Float16*)alloc(64 * 256 * 2);
    float*    ta_f  = (float*)   alloc(256 * 256 * 4);
    _Float16* ta_h  = (_Float16*)alloc(256 * 256 * 2);
    float*    sq_f  = (float*)   alloc((size_t)64 * 2000 * 4);
    float*    sa_f  = (float*)   alloc((size_t)256 * 2000 * 4);
    _Float16* pq_h  = (_Float16*)alloc((size_t)64 * 2016 * 2);
    _Float16* pa_h  = (_Float16*)alloc((size_t)256 * 2016 * 2);
    float*    oTq_f = (float*)   alloc(256 * 64 * 4);
    float*    oTa_f = (float*)   alloc(256 * 256 * 4);
    _Float16* oq_h  = (_Float16*)alloc(64 * 256 * 2);
    _Float16* oa_h  = (_Float16*)alloc(256 * 256 * 2);
    float*    r_f   = (float*)   alloc(64 * 256 * 4);

    // ---- init / weight packing (every call: ws is not preserved) ----
    zero_h16<<<(3 * 256 * 2016 + 255) / 256, 256, 0, stream>>>(Mt_h, 3 * 256 * 2016);
    zero_f32<<<(2 * 64 * 128 + 255) / 256, 256, 0, stream>>>(hbuf, 2 * 64 * 128);
    cvt_h16<<<256, 256, 0, stream>>>(U_h, U, 65536);
    cvt_h16<<<256, 256, 0, stream>>>(V_h, V, 65536);
    transpose_h16<<<256, 256, 0, stream>>>(Wt_h, W, 256, 256);
    build_WcT<<<768, 256, 0, stream>>>(WcT_h, W_ih, W_hh);
    bias_sum<<<2, 256, 0, stream>>>(bias, b_ih, b_hh);

    // ---- embeddings / memory (dominant HBM traffic) ----
    gather_M<<<6000, 256, 0, stream>>>(A_emb, memory, M_h, Mt_h);
    embed_q<<<3200, 256, 0, stream>>>(B_emb, ques, xq_h);
    embed_a<<<256, 256, 0, stream>>>(C_emb, ach, a_f, a_h);

    // ---- LSTM question encoder: 50 sequential fused steps ----
    for (int t = 0; t < 50; ++t)
        lstm_step<<<4, 256, 0, stream>>>(xq_h, WcT_h, bias, hbuf, cbuf, t);
    finalize_u<<<64, 256, 0, stream>>>(hbuf, cbuf, u_f, u_h);

    // ---- 3 memory hops ----
    for (int hop = 0; hop < 3; ++hop) {
        const _Float16* Mh  = M_h  + (size_t)hop * 2000 * 256;
        const _Float16* Mth = Mt_h + (size_t)hop * 256 * 2016;
        // question side
        gemm_abt<<<8,   256, 0, stream>>>(u_h,  U_h,  tq_f, tq_h,  64, 256, 256);
        gemm_abt<<<63,  256, 0, stream>>>(tq_h, Mh,   sq_f, nullptr, 64, 2000, 256);
        softmax_rows<<<64, 256, 0, stream>>>(sq_f, pq_h);
        gemm_abt<<<8,   256, 0, stream>>>(Mth,  pq_h, oTq_f, nullptr, 256, 64, 2016);
        add_u<<<64, 256, 0, stream>>>(u_f, u_h, oTq_f, oq_h);
        // answer side (all 4 choices flattened as 256 rows)
        gemm_abt<<<32,  256, 0, stream>>>(a_h,  V_h,  ta_f, ta_h, 256, 256, 256);
        gemm_abt<<<250, 256, 0, stream>>>(ta_h, Mh,   sa_f, nullptr, 256, 2000, 256);
        softmax_rows<<<256, 256, 0, stream>>>(sa_f, pa_h);
        gemm_abt<<<32,  256, 0, stream>>>(Mth,  pa_h, oTa_f, nullptr, 256, 256, 2016);
        add_a<<<256, 256, 0, stream>>>(a_f, a_h, oTa_f, oa_h);
    }

    // ---- prediction: r = o_q * W, pred = r·o_a, log_softmax over 4 ----
    gemm_abt<<<8, 256, 0, stream>>>(oq_h, Wt_h, r_f, nullptr, 64, 256, 256);
    predict<<<64, 128, 0, stream>>>(r_f, oa_h, out);
}
